// MambaModel_14156212207680
// MI455X (gfx1250) — matrix operand order repaired and checked by reading, not compiled, not run on hardware
//
#include <hip/hip_runtime.h>
#include <hip/hip_bf16.h>
#include <math.h>

// ---------------------------------------------------------------------------
// MambaModel forward for MI455X (gfx1250, wave32, WMMA).
// Every matmul uses the same branch-free 16x64-tile v_wmma_f32_16x16x32_f16
// kernel over pre-converted (and, where irregular, zero-padded) f16 operands:
// inner loops are pure packed b128 loads + WMMA.
// ---------------------------------------------------------------------------

typedef __attribute__((ext_vector_type(16))) _Float16 v16h;
typedef __attribute__((ext_vector_type(8)))  _Float16 v8h;
typedef __attribute__((ext_vector_type(8)))  float    v8f;

namespace cfg {
constexpr int Bsz    = 8;
constexpr int Ls     = 2048;
constexpr int Dm     = 128;    // d_model
constexpr int Hn     = 4;      // heads
constexpr int Dh     = 32;     // head dim
constexpr int Din    = 256;    // d_inner
constexpr int Dst    = 16;     // d_state
constexpr int DtR    = 8;      // dt_rank
constexpr int DbcP   = 64;     // padded dbc width (dt 0-7 | B 8-23 | C 24-39 | 0-pad)
constexpr int Layers = 6;
constexpr int Mtot   = Bsz * Ls;       // 16384 token rows
}

#define WMMA_F16(a, b, c) \
  __builtin_amdgcn_wmma_f32_16x16x32_f16(false, (a), false, (b), (short)0, (c), false, false)

// ---------------------------------------------------------------------------
// f16 operand gathers in contiguous k order (k = 16*half + s), same for A and B.
// A 16x32: lane = row (lane&15); lane-half selects K 0-15 / 16-31 -> one 32B
// contiguous run (two 16B loads).
// B 32x16: lane = col (lane&15); lane-half selects K 0-15 / 16-31 -> one 32B
// contiguous run (two 16B loads).
// ---------------------------------------------------------------------------
__device__ __forceinline__ v16h cat8(v8h lo, v8h hi) {
  return __builtin_shufflevector(lo, hi, 0, 1, 2, 3, 4, 5, 6, 7,
                                 8, 9, 10, 11, 12, 13, 14, 15);
}

__device__ __forceinline__ v16h gather_a16(const _Float16* A, int lda, int lane) {
  const int m  = lane & 15;
  const int kb = (lane >> 4) << 4;  // 0 or 16
  const _Float16* p = A + (size_t)m * lda + kb;
  return cat8(*(const v8h*)p, *(const v8h*)(p + 8));
}

__device__ __forceinline__ v16h gather_b16(const _Float16* W, int ldw, int lane) {
  const int n  = lane & 15;
  const int kb = (lane >> 4) << 4;  // 0 or 16
  const _Float16* p = W + (size_t)n * ldw + kb;
  return cat8(*(const v8h*)p, *(const v8h*)(p + 8));
}

// B 32x16 where the source is row-major over k (V tile): B[k][n] = Vb[k*ldv+n]
__device__ __forceinline__ v16h gather_bk16(const _Float16* Vb, int ldv, int lane) {
  v16h r;
  const int n  = lane & 15;
  const int kb = (lane >> 4) << 4;
#pragma unroll
  for (int v = 0; v < 8; ++v) {
    const int k = kb + (v << 1);
    r[2 * v]     = Vb[(size_t)k * ldv + n];
    r[2 * v + 1] = Vb[(size_t)(k + 1) * ldv + n];
  }
  return r;
}

// ---------------------------------------------------------------------------
// Fast aligned GEMM: C[M,N] = act(A[M,K] @ W[N,K]^T + bias), f16 operands.
// Requires M%16==0, N%64==0, K%32==0. One wave computes a 16x64 tile
// (one A gather feeds 4 WMMAs per K-step).
// ACT: 0 none, 1 relu, 2 softplus.
// ---------------------------------------------------------------------------
template <int ACT, bool OUT16>
__global__ __launch_bounds__(32) void gemm16_kernel(
    const _Float16* __restrict__ A, int lda,
    const _Float16* __restrict__ W,           // [N,K] row-major, ldw == K
    const float* __restrict__ bias,
    float* __restrict__ C, _Float16* __restrict__ C16, int ldc,
    int M, int N, int K) {
  const int lane = threadIdx.x;
  const int m0 = blockIdx.x << 4;
  const int n0 = blockIdx.y << 6;
  v8f acc[4] = {};
  for (int k0 = 0; k0 < K; k0 += 32) {
    const v16h a = gather_a16(A + (size_t)m0 * lda + k0, lda, lane);
#pragma unroll
    for (int t = 0; t < 4; ++t) {
      const v16h b = gather_b16(W + (size_t)(n0 + 16 * t) * K + k0, K, lane);
      acc[t] = WMMA_F16(a, b, acc[t]);
    }
  }
  const int n  = lane & 15;
  const int mo = (lane >> 4) << 3;
#pragma unroll
  for (int t = 0; t < 4; ++t) {
    const int cn = n0 + 16 * t + n;
    const float bv = bias ? bias[cn] : 0.f;
#pragma unroll
    for (int r = 0; r < 8; ++r) {
      float v = acc[t][r] + bv;
      if (ACT == 1)      v = fmaxf(v, 0.f);
      else if (ACT == 2) v = (v > 20.f) ? v : log1pf(expf(v));
      const size_t off = (size_t)(m0 + r + mo) * ldc + cn;
      if (OUT16) C16[off] = (_Float16)v;
      else       C[off]   = v;
    }
  }
}

// ---------------------------------------------------------------------------
// Fused flash attention over f16 qkv: one wave per (b, head, 16-query tile).
// dh=32 -> one WMMA per 16x16 score tile; online softmax; P staged through a
// 1 KB f16 LDS tile to re-layout C->A for the PV WMMAs.
// ---------------------------------------------------------------------------
__global__ __launch_bounds__(32) void attn_kernel(const _Float16* __restrict__ qkv,
                                                  _Float16* __restrict__ o) {
  using namespace cfg;
  __shared__ _Float16 pt[16 * 32];
  const int lane = threadIdx.x;
  const int q0   = blockIdx.x << 4;
  const int head = blockIdx.y;
  const int b    = blockIdx.z;

  const _Float16* base = qkv + (size_t)b * Ls * (3 * Dm);
  const _Float16* Qp = base + head * Dh;
  const _Float16* Kp = base + Dm + head * Dh;
  const _Float16* Vp = base + 2 * Dm + head * Dh;

  const float scale = 0.17677669529663687f;  // 1/sqrt(32)
  const v16h aq = gather_a16(Qp + (size_t)q0 * (3 * Dm), 3 * Dm, lane);

  float m[8], lsum[8];
#pragma unroll
  for (int r = 0; r < 8; ++r) { m[r] = -1e30f; lsum[r] = 0.f; }
  v8f acc0 = {}; v8f acc1 = {};

  const int col    = lane & 15;
  const int rowoff = (lane >> 4) << 3;

  for (int kt = 0; kt < Ls; kt += 32) {
    const v16h bk0 = gather_b16(Kp + (size_t)kt * (3 * Dm),        3 * Dm, lane);
    const v16h bk1 = gather_b16(Kp + (size_t)(kt + 16) * (3 * Dm), 3 * Dm, lane);
    v8f z0 = {}; v8f z1 = {};
    v8f s0 = WMMA_F16(aq, bk0, z0);
    v8f s1 = WMMA_F16(aq, bk1, z1);

    float sc[8];
#pragma unroll
    for (int r = 0; r < 8; ++r) {
      s0[r] *= scale;
      s1[r] *= scale;
      float t = fmaxf(s0[r], s1[r]);
      t = fmaxf(t, __shfl_xor(t, 1, 16));
      t = fmaxf(t, __shfl_xor(t, 2, 16));
      t = fmaxf(t, __shfl_xor(t, 4, 16));
      t = fmaxf(t, __shfl_xor(t, 8, 16));
      const float mn = fmaxf(m[r], t);
      sc[r] = expf(m[r] - mn);
      m[r]  = mn;
    }
#pragma unroll
    for (int r = 0; r < 8; ++r) {
      const float p0 = expf(s0[r] - m[r]);
      const float p1 = expf(s1[r] - m[r]);
      lsum[r] = lsum[r] * sc[r] + p0 + p1;
      acc0[r] *= sc[r];
      acc1[r] *= sc[r];
      pt[(r + rowoff) * 32 + col]      = (_Float16)p0;
      pt[(r + rowoff) * 32 + col + 16] = (_Float16)p1;
    }
    __syncthreads();
    const v16h pa  = gather_a16(pt, 32, lane);
    const v16h bv0 = gather_bk16(Vp + (size_t)kt * (3 * Dm),      3 * Dm, lane);
    const v16h bv1 = gather_bk16(Vp + (size_t)kt * (3 * Dm) + 16, 3 * Dm, lane);
    acc0 = WMMA_F16(pa, bv0, acc0);
    acc1 = WMMA_F16(pa, bv1, acc1);
    __syncthreads();
  }

#pragma unroll
  for (int r = 0; r < 8; ++r) {
    float t = lsum[r];
    t += __shfl_xor(t, 1, 16);
    t += __shfl_xor(t, 2, 16);
    t += __shfl_xor(t, 4, 16);
    t += __shfl_xor(t, 8, 16);
    lsum[r] = t;
  }
#pragma unroll
  for (int r = 0; r < 8; ++r) {
    const int   qrow = q0 + r + rowoff;
    const float inv  = 1.f / lsum[r];
    _Float16* op = o + ((size_t)b * Ls + qrow) * Dm + head * Dh;
    op[col]      = (_Float16)(acc0[r] * inv);
    op[col + 16] = (_Float16)(acc1[r] * inv);
  }
}

// ---------------------------------------------------------------------------
// elementwise f32 -> f16, and zero-padded matrix conversion
// dst[r*dstK + k] = (r < srcRows && k < srcK) ? src[r*srcK + k] : 0
// ---------------------------------------------------------------------------
__global__ void cvt16_kernel(const float* __restrict__ s, _Float16* __restrict__ d, int n) {
  const int i = blockIdx.x * 256 + threadIdx.x;
  if (i < n) d[i] = (_Float16)s[i];
}

__global__ void cvt16_pad_kernel(const float* __restrict__ s, _Float16* __restrict__ d,
                                 int srcRows, int srcK, int dstRows, int dstK) {
  const int i = blockIdx.x * 256 + threadIdx.x;
  if (i >= dstRows * dstK) return;
  const int r = i / dstK;
  const int k = i - r * dstK;
  d[i] = (r < srcRows && k < srcK) ? (_Float16)s[r * srcK + k] : (_Float16)0.f;
}

// ---------------------------------------------------------------------------
// h = mlp1(x) + positional encoding  (writes f32 + f16 copies)
// ---------------------------------------------------------------------------
__global__ void embed_kernel(const float* __restrict__ x, const float* __restrict__ w,
                             const float* __restrict__ bvec,
                             float* __restrict__ h, _Float16* __restrict__ h16) {
  using namespace cfg;
  const int idx = blockIdx.x * blockDim.x + threadIdx.x;
  if (idx >= Mtot * Dm) return;
  const int d  = idx & (Dm - 1);
  const int bl = idx >> 7;
  const int l  = bl & (Ls - 1);
  const float pos = (float)l;
  const float div = expf((float)(d & ~1) * (-9.210340371976184f / (float)Dm));
  const float pe  = (d & 1) ? cosf(pos * div) : sinf(pos * div);
  const float v = x[bl] * w[d] + bvec[d] + pe;
  h[idx]   = v;
  h16[idx] = (_Float16)v;
}

// ---------------------------------------------------------------------------
// out = LayerNorm(a + delta) * s + b  (f32 residual + f16 GEMM operand)
// ---------------------------------------------------------------------------
__global__ __launch_bounds__(128) void add_ln_kernel(
    const float* __restrict__ a, const float* __restrict__ delta,
    const float* __restrict__ s, const float* __restrict__ bvec,
    float* __restrict__ out, _Float16* __restrict__ out16) {
  using namespace cfg;
  __shared__ float sh[Dm];
  const int row = blockIdx.x;
  const int t   = threadIdx.x;
  const float v = a[(size_t)row * Dm + t] + delta[(size_t)row * Dm + t];
  sh[t] = v; __syncthreads();
  for (int o = 64; o > 0; o >>= 1) { if (t < o) sh[t] += sh[t + o]; __syncthreads(); }
  const float mean = sh[0] * (1.f / Dm);
  __syncthreads();
  const float dv = v - mean;
  sh[t] = dv * dv; __syncthreads();
  for (int o = 64; o > 0; o >>= 1) { if (t < o) sh[t] += sh[t + o]; __syncthreads(); }
  const float var = sh[0] * (1.f / Dm);
  const float r = dv * rsqrtf(var + 1e-5f) * s[t] + bvec[t];
  out[(size_t)row * Dm + t]   = r;
  out16[(size_t)row * Dm + t] = (_Float16)r;
}

// ---------------------------------------------------------------------------
// causal depthwise conv (width 4) + SiLU (writes f32 for scan, f16 for GEMM)
// ---------------------------------------------------------------------------
__global__ void conv_silu_kernel(const float* __restrict__ xz,
                                 const float* __restrict__ cw,
                                 const float* __restrict__ cb,
                                 float* __restrict__ xm, _Float16* __restrict__ xm16) {
  using namespace cfg;
  const int idx = blockIdx.x * blockDim.x + threadIdx.x;
  if (idx >= Mtot * Din) return;
  const int d  = idx & (Din - 1);
  const int bl = idx >> 8;
  const int l  = bl & (Ls - 1);
  const int b  = bl >> 11;
  float s = cb[d];
#pragma unroll
  for (int j = 0; j < 4; ++j) {
    const int li = l - 3 + j;
    if (li >= 0) s += cw[d * 4 + j] * xz[((size_t)b * Ls + li) * (2 * Din) + d];
  }
  const float v = s / (1.f + expf(-s));
  xm[idx]   = v;
  xm16[idx] = (_Float16)v;
}

// ---------------------------------------------------------------------------
// Selective scan: 16 lanes per (b, channel), one lane per state; y reduced
// over states with width-16 shuffles; z-gate + D-skip fused; f16 output.
// B/C read from the padded f16 dbc buffer (width DbcP).
// ---------------------------------------------------------------------------
__global__ __launch_bounds__(256) void scan_kernel(
    const float* __restrict__ dtb, const float* __restrict__ xm,
    const _Float16* __restrict__ dbc16, const float* __restrict__ xz,
    const float* __restrict__ A_log, const float* __restrict__ Dskip,
    _Float16* __restrict__ y16) {
  using namespace cfg;
  const int s    = threadIdx.x & 15;
  const int g    = threadIdx.x >> 4;
  const int pair = blockIdx.x * 16 + g;   // 0 .. B*Din-1
  const int b = pair >> 8;
  const int d = pair & (Din - 1);
  const float An = -expf(A_log[d * Dst + s]);
  const float Dv = Dskip[d];
  float st = 0.f;
  for (int l = 0; l < Ls; ++l) {
    const size_t tok = (size_t)b * Ls + l;
    const float dt = dtb[tok * Din + d];
    const float xv = xm[tok * Din + d];
    const float Bv = (float)dbc16[tok * DbcP + DtR + s];
    const float Cv = (float)dbc16[tok * DbcP + DtR + Dst + s];
    st = st * expf(dt * An) + dt * Bv * xv;
    float part = st * Cv;
    part += __shfl_xor(part, 1, 16);
    part += __shfl_xor(part, 2, 16);
    part += __shfl_xor(part, 4, 16);
    part += __shfl_xor(part, 8, 16);
    if (s == 0) {
      const float zv = xz[tok * (2 * Din) + Din + d];
      y16[tok * Din + d] = (_Float16)((part + xv * Dv) * (zv / (1.f + expf(-zv))));
    }
  }
}

// ---------------------------------------------------------------------------
// mean-pool over L, then the 2-layer head
// ---------------------------------------------------------------------------
__global__ __launch_bounds__(128) void pool_kernel(const float* __restrict__ y2,
                                                   float* __restrict__ pooled) {
  using namespace cfg;
  const int b = blockIdx.x, d = threadIdx.x;
  float s = 0.f;
  for (int l = 0; l < Ls; ++l) s += y2[((size_t)b * Ls + l) * Dm + d];
  pooled[b * Dm + d] = s * (1.f / (float)Ls);
}

__global__ __launch_bounds__(32) void head_kernel(
    const float* __restrict__ pooled,
    const float* __restrict__ w1, const float* __restrict__ b1,
    const float* __restrict__ w2, const float* __restrict__ b2,
    float* __restrict__ out) {
  using namespace cfg;
  const int b = blockIdx.x, j = threadIdx.x;
  float s = b1[j];
  for (int k = 0; k < Dm; ++k) s += pooled[b * Dm + k] * w1[j * Dm + k];
  s = fmaxf(s, 0.f) * w2[j];
  for (int off = 16; off; off >>= 1) s += __shfl_xor(s, off, 32);
  if (j == 0) out[b] = s + b2[0];
}

// ---------------------------------------------------------------------------
extern "C" void kernel_launch(void* const* d_in, const int* in_sizes, int n_in,
                              void* d_out, int out_size, void* d_ws, size_t ws_size,
                              hipStream_t stream) {
  using namespace cfg;
  const float* x          = (const float*)d_in[0];
  const float* mlp1_w     = (const float*)d_in[1];
  const float* mlp1_b     = (const float*)d_in[2];
  const float* qkv_w      = (const float*)d_in[3];
  const float* qkv_b      = (const float*)d_in[4];
  const float* attn_out_w = (const float*)d_in[5];
  const float* attn_out_b = (const float*)d_in[6];
  const float* ln1_s      = (const float*)d_in[7];
  const float* ln1_b      = (const float*)d_in[8];
  const float* ffw_w1     = (const float*)d_in[9];
  const float* ffw_b1     = (const float*)d_in[10];
  const float* ffw_w2     = (const float*)d_in[11];
  const float* ffw_b2     = (const float*)d_in[12];
  const float* ln2_s      = (const float*)d_in[13];
  const float* ln2_b      = (const float*)d_in[14];
  const float* in_proj_w  = (const float*)d_in[15];
  const float* conv_w     = (const float*)d_in[16];
  const float* conv_b     = (const float*)d_in[17];
  const float* x_proj_w   = (const float*)d_in[18];
  const float* dt_proj_w  = (const float*)d_in[19];
  const float* dt_proj_b  = (const float*)d_in[20];
  const float* A_log      = (const float*)d_in[21];
  const float* D_skip     = (const float*)d_in[22];
  const float* out_proj_w = (const float*)d_in[23];
  const float* mlp2_w1    = (const float*)d_in[24];
  const float* mlp2_b1    = (const float*)d_in[25];
  const float* mlp2_w2    = (const float*)d_in[26];
  const float* mlp2_b2    = (const float*)d_in[27];

  // -------- workspace carving --------
  const size_t BL = (size_t)Bsz * Ls;
  float* wsf = (float*)d_ws;
  float* h      = wsf;  wsf += BL * Dm;       // transformer activations (f32)
  float* t2     = wsf;  wsf += BL * Dm;       // pre-LN delta
  float* xzb    = wsf;  wsf += BL * 2 * Din;  // in_proj result
  float* xmb    = wsf;  wsf += BL * Din;      // conv+silu (f32 for scan)
  float* dtbb   = wsf;  wsf += BL * Din;      // softplus(dt)
  float* y2b    = wsf;  wsf += BL * Dm;       // out_proj result
  float* pooled = wsf;  wsf += (size_t)Bsz * Dm;

  uintptr_t up = ((uintptr_t)wsf + 15) & ~(uintptr_t)15;  // 16B align f16 pool
  _Float16* wsh = (_Float16*)up;
  _Float16* h16    = wsh;  wsh += BL * Dm;
  _Float16* qkv16  = wsh;  wsh += BL * 3 * Dm;
  _Float16* t1h    = wsh;  wsh += BL * Dm;     // attn out / ffw hidden (f16)
  _Float16* xm16   = wsh;  wsh += BL * Din;
  _Float16* dbc16  = wsh;  wsh += BL * DbcP;   // padded x_proj output
  _Float16* y16    = wsh;  wsh += BL * Din;
  _Float16* qkvw16 = wsh;  wsh += (size_t)Layers * 3 * Dm * Dm;
  _Float16* aow16  = wsh;  wsh += (size_t)Layers * Dm * Dm;
  _Float16* f1w16  = wsh;  wsh += (size_t)Layers * Dm * Dm;
  _Float16* f2w16  = wsh;  wsh += (size_t)Layers * Dm * Dm;
  _Float16* ipw16  = wsh;  wsh += (size_t)2 * Din * Dm;
  _Float16* xpw16  = wsh;  wsh += (size_t)DbcP * Din;     // 64x256, rows 40.. zero
  _Float16* dtw16  = wsh;  wsh += (size_t)Din * 32;       // 256x32, k 8.. zero
  _Float16* opw16  = wsh;  wsh += (size_t)Dm * Din;

  const int MT = Mtot;
  const dim3 blk32(32), blk128(128), blk256(256);
  auto cvt = [&](const float* s, _Float16* d, int n) {
    cvt16_kernel<<<(n + 255) / 256, blk256, 0, stream>>>(s, d, n);
  };

  // -------- one-time weight conversion (f16, zero-padded where irregular) --
  cvt(qkv_w,      qkvw16, Layers * 3 * Dm * Dm);
  cvt(attn_out_w, aow16,  Layers * Dm * Dm);
  cvt(ffw_w1,     f1w16,  Layers * Dm * Dm);
  cvt(ffw_w2,     f2w16,  Layers * Dm * Dm);
  cvt(in_proj_w,  ipw16,  2 * Din * Dm);
  cvt(out_proj_w, opw16,  Dm * Din);
  cvt16_pad_kernel<<<(DbcP * Din + 255) / 256, blk256, 0, stream>>>(
      x_proj_w, xpw16, DbcP - 24 /*40 rows*/, Din, DbcP, Din);
  cvt16_pad_kernel<<<(Din * 32 + 255) / 256, blk256, 0, stream>>>(
      dt_proj_w, dtw16, Din, DtR, Din, 32);

  // -------- embed --------
  embed_kernel<<<(MT * Dm + 255) / 256, blk256, 0, stream>>>(x, mlp1_w, mlp1_b, h, h16);

  // -------- 6 transformer layers --------
  for (int i = 0; i < Layers; ++i) {
    gemm16_kernel<0, true><<<dim3(MT / 16, (3 * Dm) / 64), blk32, 0, stream>>>(
        h16, Dm, qkvw16 + (size_t)i * 3 * Dm * Dm, qkv_b + (size_t)i * 3 * Dm,
        nullptr, qkv16, 3 * Dm, MT, 3 * Dm, Dm);

    attn_kernel<<<dim3(Ls / 16, Hn, Bsz), blk32, 0, stream>>>(qkv16, t1h);

    gemm16_kernel<0, false><<<dim3(MT / 16, Dm / 64), blk32, 0, stream>>>(
        t1h, Dm, aow16 + (size_t)i * Dm * Dm, attn_out_b + (size_t)i * Dm,
        t2, nullptr, Dm, MT, Dm, Dm);

    add_ln_kernel<<<MT, blk128, 0, stream>>>(h, t2, ln1_s + (size_t)i * Dm,
                                             ln1_b + (size_t)i * Dm, h, h16);

    gemm16_kernel<1, true><<<dim3(MT / 16, Dm / 64), blk32, 0, stream>>>(
        h16, Dm, f1w16 + (size_t)i * Dm * Dm, ffw_b1 + (size_t)i * Dm,
        nullptr, t1h, Dm, MT, Dm, Dm);

    gemm16_kernel<0, false><<<dim3(MT / 16, Dm / 64), blk32, 0, stream>>>(
        t1h, Dm, f2w16 + (size_t)i * Dm * Dm, ffw_b2 + (size_t)i * Dm,
        t2, nullptr, Dm, MT, Dm, Dm);

    add_ln_kernel<<<MT, blk128, 0, stream>>>(h, t2, ln2_s + (size_t)i * Dm,
                                             ln2_b + (size_t)i * Dm, h, h16);
  }

  // -------- Mamba block --------
  gemm16_kernel<0, false><<<dim3(MT / 16, (2 * Din) / 64), blk32, 0, stream>>>(
      h16, Dm, ipw16, nullptr, xzb, nullptr, 2 * Din, MT, 2 * Din, Dm);

  conv_silu_kernel<<<(MT * Din + 255) / 256, blk256, 0, stream>>>(
      xzb, conv_w, conv_b, xmb, xm16);

  // x_proj: N padded 40 -> 64 (zero rows), output width DbcP
  gemm16_kernel<0, true><<<dim3(MT / 16, DbcP / 64), blk32, 0, stream>>>(
      xm16, Din, xpw16, nullptr, nullptr, dbc16, DbcP, MT, DbcP, Din);

  // dt_proj: K padded 8 -> 32 (zero cols in W kill the B/C columns A reads)
  gemm16_kernel<2, false><<<dim3(MT / 16, Din / 64), blk32, 0, stream>>>(
      dbc16, DbcP, dtw16, dt_proj_b, dtbb, nullptr, Din, MT, Din, 32);

  scan_kernel<<<(Bsz * Din) / 16, blk256, 0, stream>>>(dtbb, xmb, dbc16, xzb,
                                                       A_log, D_skip, y16);

  gemm16_kernel<0, false><<<dim3(MT / 16, Dm / 64), blk32, 0, stream>>>(
      y16, Din, opw16, nullptr, y2b, nullptr, Dm, MT, Dm, Din);

  // -------- pool + head --------
  pool_kernel<<<Bsz, blk128, 0, stream>>>(y2b, pooled);
  head_kernel<<<Bsz, blk32, 0, stream>>>(pooled, mlp2_w1, mlp2_b1, mlp2_w2, mlp2_b2,
                                         (float*)d_out);
}
